// GCN_18640158065246
// MI455X (gfx1250) — matrix operand order repaired and checked by reading, not compiled, not run on hardware
//
#include <hip/hip_runtime.h>

typedef __attribute__((ext_vector_type(16))) _Float16 v16h;
typedef __attribute__((ext_vector_type(8)))  _Float16 v8h;
typedef __attribute__((ext_vector_type(8)))  float    v8f;
typedef __attribute__((ext_vector_type(4)))  int      v4i;

#define FIN 256           // input features  (reference: fin = 256)
#define HID 128           // hidden features (reference: hid = 128)
#define SPAD (FIN + 8)    // LDS row stride in halves: 528B -> conflict-free column reads

#if defined(__HIP_DEVICE_COMPILE__) && \
    __has_builtin(__builtin_amdgcn_global_load_async_to_lds_b128) && \
    __has_builtin(__builtin_amdgcn_s_wait_asynccnt)
#define USE_ASYNC_LDS 1
#else
#define USE_ASYNC_LDS 0
#endif

// ---------------- degree / normalization ----------------

__global__ void gcn_init_deg(float* __restrict__ deg, int n) {
  int i = blockIdx.x * blockDim.x + threadIdx.x;
  if (i < n) deg[i] = 1.0f;                 // self-loop contributes 1
}

__global__ void gcn_count_deg(const int* __restrict__ dst, float* __restrict__ deg, int e) {
  int i = blockIdx.x * blockDim.x + threadIdx.x;
  if (i < e) atomicAdd(&deg[dst[i]], 1.0f);
}

__global__ void gcn_make_dinv(float* __restrict__ deg, int n) {
  int i = blockIdx.x * blockDim.x + threadIdx.x;
  if (i < n) deg[i] = rsqrtf(deg[i]);       // deg >= 1 always (self-loop)
}

// ---------------- W1 fp32 -> f16 (done once; B operand is re-read 6250x) ----------------

__global__ void gcn_cvt_w1(const float* __restrict__ W1, _Float16* __restrict__ W1h, int count) {
  int i = blockIdx.x * blockDim.x + threadIdx.x;
  if (i < count) W1h[i] = (_Float16)W1[i];
}

// ---------------- layer-1 GEMM: h = x @ W1^T (WMMA f16 -> f32) ----------------
// Block = 8 waves = 8 M-tiles. W1 (f16) staged once per block into LDS
// (row stride SPAD halves), then each wave runs 8 K-steps x 8 N-tiles of
// v_wmma_f32_16x16x32_f16 with 8 v8f accumulators.
// A (16-bit 16x32): grp = lane>>4, row = lane&15;
//   a[j]   = x[row][kt*32 + 16*grp + j], a[8+j] = x[row][kt*32 + 16*grp + 8 + j]
// B (32x16, B[k][n] = W1[n][k]): b[j] = W1[nt*16 + (lane&15)][kt*32 + 16*grp + j]
// D (f32 16x16): element r -> row (r + 8*grp), col = lane&15.

__global__ void gcn_gemm1_wmma(const float* __restrict__ x,
                               const _Float16* __restrict__ W1h,
                               float* __restrict__ h, int mtiles) {
  extern __shared__ char smem_raw[];
  _Float16* sW = (_Float16*)smem_raw;       // [HID][SPAD]

  // cooperative stage: all 256 threads, 16B chunks (4096 chunks total)
  for (int c = threadIdx.x; c < (HID * FIN) / 8; c += 256) {
    const int r  = c >> 5;                  // row (FIN/8 = 32 chunks per row)
    const int cc = c & 31;
    const _Float16* g = W1h + (size_t)r * FIN + cc * 8;
    _Float16* l = sW + r * SPAD + cc * 8;
#if USE_ASYNC_LDS
    __builtin_amdgcn_global_load_async_to_lds_b128(
        (__attribute__((address_space(1))) v4i*)g,
        (__attribute__((address_space(3))) v4i*)l, 0, 0);
#else
    *(uint4*)l = *(const uint4*)g;
#endif
  }
#if USE_ASYNC_LDS
  __builtin_amdgcn_s_wait_asynccnt(0);
#endif
  __syncthreads();

  const int mt = blockIdx.x * 8 + (threadIdx.x >> 5);
  if (mt >= mtiles) return;                 // wave-uniform: EXEC stays all-ones
  const int lane = threadIdx.x & 31;
  const int grp  = lane >> 4;
  const int l16  = lane & 15;
  const size_t row = (size_t)mt * 16 + l16;

  v8f acc[HID / 16];
#pragma unroll
  for (int i = 0; i < HID / 16; ++i)
    acc[i] = (v8f){0.f, 0.f, 0.f, 0.f, 0.f, 0.f, 0.f, 0.f};

  for (int kt = 0; kt < FIN / 32; ++kt) {
    const float* xp = x + row * FIN + kt * 32 + grp * 16;
    v16h a;
#pragma unroll
    for (int j = 0; j < 8; ++j) a[j]     = (_Float16)xp[j];
#pragma unroll
    for (int j = 0; j < 8; ++j) a[8 + j] = (_Float16)xp[8 + j];

#pragma unroll
    for (int nt = 0; nt < HID / 16; ++nt) {
      const _Float16* wp = sW + (nt * 16 + l16) * SPAD + kt * 32 + grp * 16;
      const v8h b0 = *(const v8h*)(wp);     // 16B-aligned ds_load_b128
      const v8h b1 = *(const v8h*)(wp + 8);
      v16h b;
#pragma unroll
      for (int j = 0; j < 8; ++j) { b[j] = b0[j]; b[8 + j] = b1[j]; }
      acc[nt] = __builtin_amdgcn_wmma_f32_16x16x32_f16(
          /*neg_a=*/false, a, /*neg_b=*/false, b,
          /*c_mod=*/(short)0, acc[nt], /*reuse_a=*/false, /*reuse_b=*/false);
    }
  }

#pragma unroll
  for (int nt = 0; nt < HID / 16; ++nt) {
#pragma unroll
    for (int r = 0; r < 8; ++r) {
      h[((size_t)mt * 16 + r + 8 * grp) * HID + nt * 16 + l16] = acc[nt][r];
    }
  }
}

// scalar tail for rows not covered by full 16-row tiles (N=100000 -> none, but be safe)
__global__ void gcn_gemm1_tail(const float* __restrict__ x, const float* __restrict__ W1,
                               float* __restrict__ h, int row0, int n) {
  int t = blockIdx.x * blockDim.x + threadIdx.x;
  int row = row0 + t / HID;
  int f = t % HID;
  if (row >= n) return;
  float s = 0.f;
  for (int k = 0; k < FIN; ++k) s += x[(size_t)row * FIN + k] * W1[(size_t)f * FIN + k];
  h[(size_t)row * HID + f] = s;
}

// ---------------- layer-1 aggregation ----------------

__global__ void gcn_selfloop1(const float* __restrict__ h, const float* __restrict__ dinv,
                              float* __restrict__ agg, int n) {
  size_t idx = (size_t)blockIdx.x * blockDim.x + threadIdx.x;
  if (idx >= (size_t)n * HID) return;
  int node = (int)(idx >> 7);
  float d = dinv[node];
  agg[idx] = h[idx] * d * d;
}

// one wave per edge (wave-uniform edge index -> scalar loads), 4 features per lane
__global__ void gcn_scatter1(const float* __restrict__ h, const float* __restrict__ dinv,
                             const int* __restrict__ src, const int* __restrict__ dst,
                             float* __restrict__ agg, int e) {
  int edge = blockIdx.x * 8 + __builtin_amdgcn_readfirstlane((int)(threadIdx.x >> 5));
  if (edge >= e) return;
  int lane = threadIdx.x & 31;
  int s = src[edge], d = dst[edge];
  float w = dinv[s] * dinv[d];
  const float4 v = *reinterpret_cast<const float4*>(h + (size_t)s * HID + lane * 4);
  float* ap = agg + (size_t)d * HID + lane * 4;
  atomicAdd(ap + 0, v.x * w);
  atomicAdd(ap + 1, v.y * w);
  atomicAdd(ap + 2, v.z * w);
  atomicAdd(ap + 3, v.w * w);
}

__global__ void gcn_bias_relu(float* __restrict__ agg, const float* __restrict__ b1, int n) {
  size_t idx = (size_t)blockIdx.x * blockDim.x + threadIdx.x;
  if (idx >= (size_t)n * HID) return;
  float v = agg[idx] + b1[idx & (HID - 1)];
  agg[idx] = v > 0.f ? v : 0.f;
}

// ---------------- layer 2: h2 = agg @ W2^T, out = self-loop + bias ----------------

__global__ void gcn_layer2(const float* __restrict__ agg, const float* __restrict__ W2,
                           const float* __restrict__ b2, const float* __restrict__ dinv,
                           float* __restrict__ h2, float* __restrict__ out, int n) {
  int i = blockIdx.x * blockDim.x + threadIdx.x;
  if (i >= n) return;
  const float4* a4 = reinterpret_cast<const float4*>(agg + (size_t)i * HID);
  const float4* w4 = reinterpret_cast<const float4*>(W2);
  float s = 0.f;
#pragma unroll
  for (int j = 0; j < HID / 4; ++j) {
    float4 a = a4[j], w = w4[j];
    s += a.x * w.x + a.y * w.y + a.z * w.z + a.w * w.w;
  }
  h2[i] = s;
  float d = dinv[i];
  out[i] = s * d * d + b2[0];               // fully initializes d_out before scatter
}

__global__ void gcn_scatter2(const float* __restrict__ h2, const float* __restrict__ dinv,
                             const int* __restrict__ src, const int* __restrict__ dst,
                             float* __restrict__ out, int e) {
  int i = blockIdx.x * blockDim.x + threadIdx.x;
  if (i >= e) return;
  int s = src[i], d = dst[i];
  atomicAdd(&out[d], h2[s] * dinv[s] * dinv[d]);
}

// ---------------- host launch ----------------

extern "C" void kernel_launch(void* const* d_in, const int* in_sizes, int n_in,
                              void* d_out, int out_size, void* d_ws, size_t ws_size,
                              hipStream_t stream) {
  const float* x  = (const float*)d_in[0];
  const int*   ei = (const int*)d_in[1];
  const float* W1 = (const float*)d_in[2];
  const float* b1 = (const float*)d_in[3];
  const float* W2 = (const float*)d_in[4];
  const float* b2 = (const float*)d_in[5];
  float* out = (float*)d_out;

  const int n = in_sizes[0] / FIN;
  const int e = in_sizes[1] / 2;
  const int* srcv = ei;
  const int* dstv = ei + e;

  // workspace carve (floats), 256B-aligned segments
  size_t nAlign = ((size_t)n + 63) & ~(size_t)63;
  float* dinv = (float*)d_ws;                  // n         (deg -> dinv in place)
  float* h    = dinv + nAlign;                 // n * HID
  float* agg  = h + nAlign * HID;              // n * HID
  float* h2   = agg + nAlign * HID;            // n
  _Float16* w1h = (_Float16*)(h2 + nAlign);    // HID * FIN halves (64 KB)
  (void)ws_size;

  const int tb = 256;
  gcn_init_deg <<<(n + tb - 1) / tb, tb, 0, stream>>>(dinv, n);
  gcn_count_deg<<<(e + tb - 1) / tb, tb, 0, stream>>>(dstv, dinv, e);
  gcn_make_dinv<<<(n + tb - 1) / tb, tb, 0, stream>>>(dinv, n);
  gcn_cvt_w1   <<<(HID * FIN + tb - 1) / tb, tb, 0, stream>>>(W1, w1h, HID * FIN);

  const int mtiles = n / 16;
  if (mtiles > 0) {
    int blocks = (mtiles + 7) / 8;             // 8 waves (tiles) per 256-thread block
    size_t smem = (size_t)HID * SPAD * sizeof(_Float16);   // 67584 B dynamic LDS
    gcn_gemm1_wmma<<<blocks, 256, smem, stream>>>(x, w1h, h, mtiles);
  }
  const int rem = n - mtiles * 16;
  if (rem > 0) {
    int t = rem * HID;
    gcn_gemm1_tail<<<(t + tb - 1) / tb, tb, 0, stream>>>(x, W1, h, mtiles * 16, n);
  }

  const long long nf = (long long)n * HID;
  gcn_selfloop1<<<(unsigned)((nf + tb - 1) / tb), tb, 0, stream>>>(h, dinv, agg, n);
  gcn_scatter1 <<<(e + 7) / 8, 256, 0, stream>>>(h, dinv, srcv, dstv, agg, e);
  gcn_bias_relu<<<(unsigned)((nf + tb - 1) / tb), tb, 0, stream>>>(agg, b1, n);
  gcn_layer2   <<<(n + tb - 1) / tb, tb, 0, stream>>>(agg, W2, b2, dinv, h2, out, n);
  gcn_scatter2 <<<(e + tb - 1) / tb, tb, 0, stream>>>(h2, dinv, srcv, dstv, out, e);
}
